// GCNEncoder_27633819583001
// MI455X (gfx1250) — matrix operand order repaired and checked by reading, not compiled, not run on hardware
//
#include <hip/hip_runtime.h>

// ---------------------------------------------------------------------------
// 2-layer GCN (PyG GCNConv semantics) for MI455X / gfx1250, wave32.
//   layer:  h = X @ W                       (bf16 WMMA, f32 accumulate)
//           agg[d] = sum_e dinv[s]*dinv[d]*h[s]  + dinv[d]^2 * h[d]  (self loop)
//           out = relu?(agg + b)
// Aggregation dominates (~102M fp32 atomics per layer), but both 25.6MB
// feature buffers are L2-resident (192MB L2); only the edge list streams
// from HBM.  W is pre-swizzled into WMMA B-fragment order so each B operand
// is one contiguous 32B LDS read (2x ds_load_b128) instead of 16 scalar
// ds_load_u16 + merges.
// ---------------------------------------------------------------------------

typedef __attribute__((ext_vector_type(16))) __bf16 v16bf;
typedef __attribute__((ext_vector_type(8)))  float  v8f;

#define HID 128

// ---- small prep kernels ----------------------------------------------------

// Convert W (f32, row-major K x N = 128 x 128) into bf16 in WMMA B-fragment
// order: frag index f = (k/32)*8 + n/16  (k-step major, n-tile minor),
// lane = 16*(((k%32)/8)%2) + n%16, element j = k%8 + 8*((k%32)/16).
// Layout: Wf[((f*32) + lane)*16 + j]
__global__ void wconv_kernel(const float* __restrict__ W, __bf16* __restrict__ Wf) {
    int i = blockIdx.x * blockDim.x + threadIdx.x;
    if (i < HID * HID) {
        int k = i >> 7;        // row (K)
        int n = i & (HID - 1); // col (N)
        int f    = (k >> 5) * 8 + (n >> 4);
        int lane = ((k >> 3) & 1) * 16 + (n & 15);
        int j    = (k & 7) + (((k & 31) >> 4) << 3);
        Wf[((size_t)(f * 32 + lane) << 4) + j] = (__bf16)W[i];
    }
}

__global__ void deg_init_kernel(float* deg, int n) {
    int i = blockIdx.x * blockDim.x + threadIdx.x;
    if (i < n) deg[i] = 1.0f;  // self loop
}

__global__ void deg_count_kernel(const int* __restrict__ dst, float* deg, int E, int n) {
    int e = blockIdx.x * blockDim.x + threadIdx.x;
    if (e < E) {
        int d = dst[e];
        if ((unsigned)d < (unsigned)n) atomicAdd(&deg[d], 1.0f);
    }
}

__global__ void dinv_kernel(float* deg, int n) {
    int i = blockIdx.x * blockDim.x + threadIdx.x;
    if (i < n) deg[i] = rsqrtf(deg[i]);  // deg >= 1 always (self loop)
}

// ---- dense GEMM via WMMA ---------------------------------------------------
// Block = 256 threads = 8 waves; block covers 128 rows, each wave a 16-row
// strip x all 128 cols (8 tiles of 16x16), K-loop in 4 steps of 32.
// W (fragment-ordered bf16, 32KB) is staged into LDS with coalesced b128s.

__global__ __launch_bounds__(256)
void gcn_gemm_wmma_kernel(const float* __restrict__ X, const __bf16* __restrict__ Wf,
                          float* __restrict__ H, int nrows) {
    __shared__ __bf16 Wl[HID * HID];  // 32 KB of 320 KB LDS, fragment order
    {
        uint4* d = (uint4*)Wl;
        const uint4* s = (const uint4*)Wf;
        for (int i = threadIdx.x; i < (HID * HID) / 8; i += 256) d[i] = s[i];
    }
    __syncthreads();

    const int lane = threadIdx.x & 31;
    const int wave = threadIdx.x >> 5;
    const int half = lane >> 4;    // 0 | 1 : lane half selects K sub-runs
    const int l16  = lane & 15;
    const int rowBase = (blockIdx.x * 8 + wave) * 16;
    const int m = rowBase + l16;
    const bool rowOK = (m < nrows);
    const float* xrow = X + (size_t)(rowOK ? m : 0) * HID;
    const char* wbase = (const char*)Wl + (size_t)lane * 32;  // this lane's 32B slot

    v8f acc[8] = {};

    #pragma unroll
    for (int ki = 0; ki < 4; ++ki) {
        const int k0 = ki * 32;
        // A fragment, 16-bit A 16x32 layout: elem j -> K = 16*(j/8) + 8*half + (j%8)
        float4 r0 = *(const float4*)(xrow + k0 + half * 8);
        float4 r1 = *(const float4*)(xrow + k0 + half * 8 + 4);
        float4 r2 = *(const float4*)(xrow + k0 + 16 + half * 8);
        float4 r3 = *(const float4*)(xrow + k0 + 16 + half * 8 + 4);
        v16bf a;
        a[0]  = (__bf16)r0.x; a[1]  = (__bf16)r0.y; a[2]  = (__bf16)r0.z; a[3]  = (__bf16)r0.w;
        a[4]  = (__bf16)r1.x; a[5]  = (__bf16)r1.y; a[6]  = (__bf16)r1.z; a[7]  = (__bf16)r1.w;
        a[8]  = (__bf16)r2.x; a[9]  = (__bf16)r2.y; a[10] = (__bf16)r2.z; a[11] = (__bf16)r2.w;
        a[12] = (__bf16)r3.x; a[13] = (__bf16)r3.y; a[14] = (__bf16)r3.z; a[15] = (__bf16)r3.w;

        #pragma unroll
        for (int t = 0; t < 8; ++t) {
            // B fragment: one contiguous 32B LDS read per lane
            v16bf b = *(const v16bf*)(wbase + (size_t)(ki * 8 + t) * 32 * 32);
            acc[t] = __builtin_amdgcn_wmma_f32_16x16x32_bf16(
                false, a, false, b, (short)0, acc[t], false, false);
        }
    }

    // C/D layout: VGPR r, lane -> row = rowBase + r + 8*half, col = t*16 + l16
    #pragma unroll
    for (int t = 0; t < 8; ++t) {
        #pragma unroll
        for (int r = 0; r < 8; ++r) {
            int row = rowBase + r + half * 8;
            if (row < nrows) H[(size_t)row * HID + t * 16 + l16] = acc[t][r];
        }
    }
}

// ---- aggregation -----------------------------------------------------------

__global__ void self_init_kernel(const float* __restrict__ H, const float* __restrict__ dinv,
                                 float* __restrict__ agg, int n) {
    int t = blockIdx.x * blockDim.x + threadIdx.x;
    if (t < n * HID) {
        int i = t >> 7;
        float w = dinv[i] * dinv[i];
        agg[t] = H[t] * w;
    }
}

// 32 lanes per edge, each lane does a float4 gather + 4 non-returning f32 atomics.
__global__ void edge_scatter_kernel(const int* __restrict__ src, const int* __restrict__ dst,
                                    const float* __restrict__ H, const float* __restrict__ dinv,
                                    float* agg, int E, int n) {
    int t = blockIdx.x * blockDim.x + threadIdx.x;
    int e = t >> 5;
    if (e >= E) return;
    int c4 = (t & 31) << 2;
    int s = src[e], d = dst[e];
    if ((unsigned)s >= (unsigned)n || (unsigned)d >= (unsigned)n) return;
    float norm = dinv[s] * dinv[d];
    float4 hv = *(const float4*)(H + (size_t)s * HID + c4);
    float* o = agg + (size_t)d * HID + c4;
    atomicAdd(o + 0, hv.x * norm);
    atomicAdd(o + 1, hv.y * norm);
    atomicAdd(o + 2, hv.z * norm);
    atomicAdd(o + 3, hv.w * norm);
}

__global__ void bias_act_kernel(float* agg, const float* __restrict__ bias, int n, int doRelu) {
    int t = blockIdx.x * blockDim.x + threadIdx.x;
    if (t < n * HID) {
        float v = agg[t] + bias[t & (HID - 1)];
        agg[t] = doRelu ? fmaxf(v, 0.0f) : v;
    }
}

// ---- launch ---------------------------------------------------------------

extern "C" void kernel_launch(void* const* d_in, const int* in_sizes, int n_in,
                              void* d_out, int out_size, void* d_ws, size_t ws_size,
                              hipStream_t stream) {
    (void)n_in; (void)out_size; (void)ws_size;
    const float* x  = (const float*)d_in[0];
    const int*   ei = (const int*)d_in[1];   // int32 (JAX x64 disabled)
    const float* W1 = (const float*)d_in[2];
    const float* b1 = (const float*)d_in[3];
    const float* W2 = (const float*)d_in[4];
    const float* b2 = (const float*)d_in[5];

    const int N = in_sizes[0] / HID;
    const int E = in_sizes[1] / 2;
    const int* src = ei;
    const int* dst = ei + E;

    char* ws = (char*)d_ws;
    size_t off = 0;
    auto alloc = [&](size_t bytes) -> void* {
        void* p = ws + off;
        off = (off + bytes + 255) & ~(size_t)255;
        return p;
    };
    float*  dinv = (float*)alloc((size_t)N * 4);
    __bf16* W1f  = (__bf16*)alloc((size_t)HID * HID * 2);
    __bf16* W2f  = (__bf16*)alloc((size_t)HID * HID * 2);
    float*  bufH = (float*)alloc((size_t)N * HID * 4);
    float*  bufA = (float*)alloc((size_t)N * HID * 4);
    float*  out  = (float*)d_out;

    const int T = 256;
    const int wBlocks    = (HID * HID + T - 1) / T;
    const int nBlocks    = (N + T - 1) / T;
    const int eBlocks    = (E + T - 1) / T;
    const int gemmBlocks = (N + 127) / 128;
    const int elemBlocks = (int)(((size_t)N * HID + T - 1) / T);
    const int edgeBlocks = (int)(((size_t)E * 32 + T - 1) / T);

    // weights -> bf16 fragment order; degrees -> dinv
    wconv_kernel<<<wBlocks, T, 0, stream>>>(W1, W1f);
    wconv_kernel<<<wBlocks, T, 0, stream>>>(W2, W2f);
    deg_init_kernel<<<nBlocks, T, 0, stream>>>(dinv, N);
    deg_count_kernel<<<eBlocks, T, 0, stream>>>(dst, dinv, E, N);
    dinv_kernel<<<nBlocks, T, 0, stream>>>(dinv, N);

    // layer 1
    gcn_gemm_wmma_kernel<<<gemmBlocks, T, 0, stream>>>(x, W1f, bufH, N);
    self_init_kernel<<<elemBlocks, T, 0, stream>>>(bufH, dinv, bufA, N);
    edge_scatter_kernel<<<edgeBlocks, T, 0, stream>>>(src, dst, bufH, dinv, bufA, E, N);
    bias_act_kernel<<<elemBlocks, T, 0, stream>>>(bufA, b1, N, 1);

    // layer 2 (accumulate straight into d_out)
    gcn_gemm_wmma_kernel<<<gemmBlocks, T, 0, stream>>>(bufA, W2f, bufH, N);
    self_init_kernel<<<elemBlocks, T, 0, stream>>>(bufH, dinv, out, N);
    edge_scatter_kernel<<<edgeBlocks, T, 0, stream>>>(src, dst, bufH, dinv, out, E, N);
    bias_act_kernel<<<elemBlocks, T, 0, stream>>>(out, b2, N, 0);
}
